// ColorNet_19713899889090
// MI455X (gfx1250) — matrix operand order repaired and checked by reading, not compiled, not run on hardware
//
#include <hip/hip_runtime.h>
#include <stdint.h>

typedef __attribute__((ext_vector_type(16))) _Float16 v16h;
typedef __attribute__((ext_vector_type(8)))  _Float16 v8h;
typedef __attribute__((ext_vector_type(8)))  float    v8f;

#define WMMA16(a,b,c) __builtin_amdgcn_wmma_f32_16x16x32_f16(false,(a),false,(b),(short)0,(c),false,false)

// ---------------- async global->LDS staging (CDNA5 ASYNCcnt path) ----------------
// Copies `bytes` (multiple of 16) from global `gsrc` to LDS `ldsDst`, 16B per lane
// per instruction, distributed over `stride` threads. Must be followed by
// wait_async() + a barrier before any thread reads the LDS region.
__device__ __forceinline__ void async_copy_to_lds(const void* gsrc, void* ldsDst,
                                                  int bytes, int tid, int stride) {
  unsigned l0 = (unsigned)(uintptr_t)ldsDst;          // low 32 bits = LDS offset
  unsigned long long g0 = (unsigned long long)gsrc;   // uniform base -> SGPR pair
  for (int i = tid * 16; i < bytes; i += stride * 16) {
    unsigned l = l0 + i;
    asm volatile("global_load_async_to_lds_b128 %0, %1, %2"
                 :: "v"(l), "v"(i), "s"(g0) : "memory");
  }
}
__device__ __forceinline__ void wait_async() {
  asm volatile("s_wait_asynccnt 0x0" ::: "memory");
}

// ---------------- WMMA fragment loaders (from LDS, f16) ----------------
// A: 16xK row-major tile at Xs+m0*ld. 16-bit A 16x32 layout:
//   lane<16: K = k0+0..7 (v0..3) and k0+16..23 (v4..7); lane>=16: +8.
__device__ __forceinline__ v16h frag_a(const _Float16* Xs, int ld, int m0, int k0, int lane) {
  int r  = lane & 15;
  int hh = (lane >> 4) << 3;
  const _Float16* p = Xs + (m0 + r) * ld + k0 + hh;
  v8h x0 = *(const v8h*)p;
  v8h x1 = *(const v8h*)(p + 16);
  v16h a;
#pragma unroll
  for (int i = 0; i < 8; ++i) { a[i] = x0[i]; a[i + 8] = x1[i]; }
  return a;
}
// B: Wt is the weight stored n-major (Wt[n*ld + k] = W[k][n]).
//   lane n (0..15) holds K = k0+0..7 and k0+16..23; lane n+16 holds +8.
__device__ __forceinline__ v16h frag_b(const _Float16* Wt, int ld, int n0, int k0, int lane) {
  int n  = n0 + (lane & 15);
  int ko = k0 + ((lane >> 4) << 3);
  const _Float16* p = Wt + n * ld + ko;
  v8h b0 = *(const v8h*)p;
  v8h b1 = *(const v8h*)(p + 16);
  v16h b;
#pragma unroll
  for (int i = 0; i < 8; ++i) { b[i] = b0[i]; b[i + 8] = b1[i]; }
  return b;
}

// ---------------- utility kernels ----------------
__global__ void zero_f_k(float* p, int n) {
  int i = blockIdx.x * blockDim.x + threadIdx.x;
  if (i < n) p[i] = 0.f;
}
__global__ void fill_i32_k(int* p, int v, int n) {
  int i = blockIdx.x * blockDim.x + threadIdx.x;
  if (i < n) p[i] = v;
}
// transpose+convert weights to f16 n-major: out[e][n*K+k] = W[e][k*64+n]
__global__ void prep_wt_k(const float* __restrict__ W, _Float16* __restrict__ out,
                          int K, int nE) {
  int i = blockIdx.x * blockDim.x + threadIdx.x;
  int per = 64 * K;
  if (i >= per * nE) return;
  int e = i / per, r = i - e * per;
  int n = r / K, k = r - n * K;
  out[i] = (_Float16)W[(size_t)e * per + (size_t)k * 64 + n];
}

// ---------------- small embedding layers ----------------
__global__ void cons_emb1_k(const float* __restrict__ x, const float* lg, const float* lb,
                            const float* __restrict__ W1, const float* __restrict__ b1,
                            float* __restrict__ h, int N) {
  int n = blockIdx.x; if (n >= N) return;
  float a = x[n * 2], b = x[n * 2 + 1];
  float m = 0.5f * (a + b);
  float va = 0.5f * ((a - m) * (a - m) + (b - m) * (b - m));
  float rs = rsqrtf(va + 1e-5f);
  float z0 = (a - m) * rs * lg[0] + lb[0];
  float z1 = (b - m) * rs * lg[1] + lb[1];
  int k = threadIdx.x;
  float y = z0 * W1[k] + z1 * W1[64 + k] + b1[k];
  h[n * 64 + k] = fmaxf(y, 0.f);
}
__global__ void var_emb1_k(const float* __restrict__ x, const float* lg, const float* lb,
                           const float* __restrict__ W1, const float* __restrict__ b1,
                           float* __restrict__ h, int N) {
  int n = blockIdx.x; if (n >= N) return;
  float v[7]; float s = 0.f;
#pragma unroll
  for (int j = 0; j < 7; ++j) { v[j] = x[n * 7 + j]; s += v[j]; }
  float m = s * (1.f / 7.f), q = 0.f;
#pragma unroll
  for (int j = 0; j < 7; ++j) q += (v[j] - m) * (v[j] - m);
  float rs = rsqrtf(q * (1.f / 7.f) + 1e-5f);
  int k = threadIdx.x;
  float y = b1[k];
#pragma unroll
  for (int j = 0; j < 7; ++j) y += ((v[j] - m) * rs * lg[j] + lb[j]) * W1[j * 64 + k];
  h[n * 64 + k] = fmaxf(y, 0.f);
}
__global__ void grp_emb_k(const float* __restrict__ x, const float* lg, const float* lb,
                          const float* __restrict__ W, const float* __restrict__ b,
                          float* __restrict__ ge, int N) {
  __shared__ float z[32];
  int n = blockIdx.x; if (n >= N) return;
  int t = threadIdx.x;
  if (t < 32) {
    float v = x[n * 32 + t];
    float s = v, q = v * v;
#pragma unroll
    for (int m = 1; m < 32; m <<= 1) { s += __shfl_xor(s, m, 32); q += __shfl_xor(q, m, 32); }
    float mean = s * (1.f / 32.f);
    float var  = q * (1.f / 32.f) - mean * mean;
    z[t] = (v - mean) * rsqrtf(var + 1e-5f) * lg[t] + lb[t];
  }
  __syncthreads();
  float y = b[t];
#pragma unroll
  for (int j = 0; j < 32; ++j) y += z[j] * W[j * 64 + t];
  ge[n * 64 + t] = fmaxf(y, 0.f);
}

// ---------------- generic node GEMM: Y = act(X[NxK] @ W[Kx64] + bias (+ bias2*scale)) (+res)
// W16 is pre-transposed f16 (n-major, 64 x K), staged to LDS via async DMA.
__global__ void gemm_node_k(const float* __restrict__ X, int K,
                            const _Float16* __restrict__ W16,
                            const float* __restrict__ bias,
                            const float* __restrict__ bias2,
                            const float* __restrict__ bias2scale,
                            const float* __restrict__ residual,
                            float* __restrict__ Y, int N, int doRelu) {
  __shared__ __align__(16) _Float16 Wt[64 * 128];   // n-major, ld = K
  __shared__ __align__(16) _Float16 Xs[64 * 128];   // row-major, ld = K
  int tid = threadIdx.x, lane = tid & 31, wave = tid >> 5;
  async_copy_to_lds(W16, Wt, 64 * K * 2, tid, 128);   // DMA weights while we stage X
  int ksh = (K == 128) ? 7 : 6;
  int r0 = blockIdx.x * 64;
  for (int i = tid; i < 64 * K; i += 128) {
    int r = i >> ksh, k = i & (K - 1);
    int gr = r0 + r;
    Xs[i] = (gr < N) ? (_Float16)X[(size_t)gr * K + k] : (_Float16)0.f;
  }
  wait_async();
  __syncthreads();
  int m0 = wave * 16;
  v8f acc[4];
#pragma unroll
  for (int t = 0; t < 4; ++t)
#pragma unroll
    for (int v = 0; v < 8; ++v) acc[t][v] = 0.f;
  for (int kc = 0; kc < K; kc += 32) {
    v16h a = frag_a(Xs, K, m0, kc, lane);
#pragma unroll
    for (int t = 0; t < 4; ++t) {
      v16h b = frag_b(Wt, K, t * 16, kc, lane);
      acc[t] = WMMA16(a, b, acc[t]);
    }
  }
  int n = lane & 15;
  int rb = r0 + m0 + ((lane >> 4) << 3);
  float sc = bias2 ? bias2scale[0] : 0.f;
#pragma unroll
  for (int t = 0; t < 4; ++t) {
    int nn = t * 16 + n;
    float bv = bias ? bias[nn] : 0.f;
    if (bias2) bv += bias2[nn] * sc;
#pragma unroll
    for (int v = 0; v < 8; ++v) {
      int row = rb + v;
      if (row < N) {
        float y = acc[t][v] + bv;
        if (doRelu) y = fmaxf(y, 0.f);
        if (residual) y += residual[(size_t)row * 64 + nn];
        Y[(size_t)row * 64 + nn] = y;
      }
    }
  }
}

// ---------------- per-edge message kernel ----------------
// pre = R[dst] + L[src]; h = relu(LN(pre)); msg = h @ Wf + bf; atomic agg[dst] += msg.
// 128 threads = 4 waves, each wave processes 16 edges (one WMMA M-tile).
__global__ void edge_msg_k(const float* __restrict__ R, const float* __restrict__ L,
                           const int* __restrict__ srcIdx, const int* __restrict__ dstIdx,
                           const float* __restrict__ lfg, const float* __restrict__ lfb,
                           const _Float16* __restrict__ Wf16,   // pre-transposed n-major
                           const float* __restrict__ bf,
                           float* __restrict__ agg, int E) {
  __shared__ __align__(16) _Float16 Wft[64 * 64];
  __shared__ __align__(16) _Float16 Hs[4][16 * 64];
  __shared__ int dsts[4][16];
  int tid = threadIdx.x, lane = tid & 31, wave = tid >> 5;
  async_copy_to_lds(Wf16, Wft, 64 * 64 * 2, tid, 128);   // DMA Wf while doing LN math
  int e0 = (blockIdx.x * 4 + wave) * 16;
  if (e0 < E) {
    int il = lane >> 1;
    int e  = e0 + il;
    int valid = e < E;
    int ee = valid ? e : (E - 1);
    int s = srcIdx[ee], d = dstIdx[ee];
    if ((lane & 1) == 0) dsts[wave][il] = valid ? d : -1;
    int kh = (lane & 1) * 32;
    const float4* r4 = (const float4*)(R + (size_t)d * 64 + kh);
    const float4* l4 = (const float4*)(L + (size_t)s * 64 + kh);
    float x[32]; float sum = 0.f, sq = 0.f;
#pragma unroll
    for (int j = 0; j < 8; ++j) {
      float4 rv = r4[j], lv = l4[j];
      float v0 = rv.x + lv.x, v1 = rv.y + lv.y, v2 = rv.z + lv.z, v3 = rv.w + lv.w;
      x[4*j] = v0; x[4*j+1] = v1; x[4*j+2] = v2; x[4*j+3] = v3;
      sum += v0 + v1 + v2 + v3;
      sq  += v0*v0 + v1*v1 + v2*v2 + v3*v3;
    }
    sum += __shfl_xor(sum, 1, 32);
    sq  += __shfl_xor(sq,  1, 32);
    float mean = sum * (1.f / 64.f);
    float var  = sq  * (1.f / 64.f) - mean * mean;
    float rstd = rsqrtf(var + 1e-5f);
#pragma unroll
    for (int j = 0; j < 32; ++j) {
      int k = kh + j;
      float hv = (x[j] - mean) * rstd * lfg[k] + lfb[k];
      Hs[wave][il * 64 + k] = (_Float16)fmaxf(hv, 0.f);
    }
  }
  wait_async();
  __syncthreads();
  if (e0 >= E) return;
  int n = lane & 15;
  v8f acc[4];
#pragma unroll
  for (int t = 0; t < 4; ++t) {
    float bv = bf[t * 16 + n];
#pragma unroll
    for (int v = 0; v < 8; ++v) acc[t][v] = bv;   // per-edge bias folded into accumulator
  }
  for (int kc = 0; kc < 64; kc += 32) {
    v16h a = frag_a(Hs[wave], 64, 0, kc, lane);
#pragma unroll
    for (int t = 0; t < 4; ++t) {
      v16h b = frag_b(Wft, 64, t * 16, kc, lane);
      acc[t] = WMMA16(a, b, acc[t]);
    }
  }
  int rbase = (lane >> 4) << 3;
#pragma unroll
  for (int t = 0; t < 4; ++t) {
    int nn = t * 16 + n;
#pragma unroll
    for (int v = 0; v < 8; ++v) {
      int d = dsts[wave][rbase + v];
      if (d >= 0) atomicAdd(&agg[(size_t)d * 64 + nn], acc[t][v]);
    }
  }
}

// ---------------- post-aggregation: cat = [LN(agg)*g+b , right] ----------------
__global__ void postcat_k(const float* __restrict__ agg, const float* __restrict__ right,
                          const float* __restrict__ g, const float* __restrict__ b,
                          float* __restrict__ cat, int N) {
  int wave = threadIdx.x >> 5, lane = threadIdx.x & 31;
  int node = blockIdx.x * 8 + wave;
  if (node >= N) return;
  float x0 = agg[(size_t)node * 64 + lane];
  float x1 = agg[(size_t)node * 64 + 32 + lane];
  float s = x0 + x1, q = x0 * x0 + x1 * x1;
#pragma unroll
  for (int m = 1; m < 32; m <<= 1) { s += __shfl_xor(s, m, 32); q += __shfl_xor(q, m, 32); }
  float mean = s * (1.f / 64.f);
  float var  = q * (1.f / 64.f) - mean * mean;
  float rstd = rsqrtf(var + 1e-5f);
  size_t o = (size_t)node * 128;
  cat[o + lane]      = (x0 - mean) * rstd * g[lane] + b[lane];
  cat[o + 32 + lane] = (x1 - mean) * rstd * g[32 + lane] + b[32 + lane];
  cat[o + 64 + lane] = right[(size_t)node * 64 + lane];
  cat[o + 96 + lane] = right[(size_t)node * 64 + 32 + lane];
}

// ---------------- MoE dispatch ----------------
__global__ void moe_count_k(const int* __restrict__ color, int* counts, int N) {
  int i = blockIdx.x * blockDim.x + threadIdx.x;
  if (i < N) atomicAdd(&counts[color[i]], 1);
}
__global__ void moe_scan_k(const int* __restrict__ counts, int* offsets, int* padded, int* cursor) {
  if (threadIdx.x == 0) {
    int o = 0, p = 0;
    for (int e = 0; e < 32; ++e) {
      offsets[e] = o; padded[e] = p; cursor[e] = p;
      o += counts[e];
      p += (counts[e] + 15) & ~15;
    }
    offsets[32] = o; padded[32] = p;
  }
}
__global__ void moe_scatter_k(const int* __restrict__ color, int* cursor, int* idxbuf, int N) {
  int i = blockIdx.x * blockDim.x + threadIdx.x;
  if (i < N) {
    int pos = atomicAdd(&cursor[color[i]], 1);
    idxbuf[pos] = i;
  }
}
// one wave per 16-row expert tile: out[rows] = relu(cat[rows]@W1[e]+b1[e]) @ W2[e] + b2[e]
// W1_16/W2_16 are pre-transposed f16 expert weights (n-major), DMA-staged.
__global__ void color_gemm_k(const float* __restrict__ cat, const int* __restrict__ idxbuf,
                             const int* __restrict__ padded,
                             const _Float16* __restrict__ W1_16, const float* __restrict__ B1,
                             const _Float16* __restrict__ W2_16, const float* __restrict__ B2,
                             float* __restrict__ out) {
  __shared__ __align__(16) _Float16 W1t[64 * 128];
  __shared__ __align__(16) _Float16 W2t[64 * 64];
  __shared__ __align__(16) _Float16 Xs[16 * 128];
  __shared__ __align__(16) _Float16 Hs[16 * 64];
  __shared__ int ridx[16];
  int lane = threadIdx.x;
  int row0 = blockIdx.x * 16;
  if (row0 >= padded[32]) return;
  int e = 0;
  while (padded[e + 1] <= row0) ++e;
  if (lane < 16) ridx[lane] = idxbuf[row0 + lane];
  __syncthreads();
  async_copy_to_lds(W1_16 + (size_t)e * 64 * 128, W1t, 64 * 128 * 2, lane, 32);
  async_copy_to_lds(W2_16 + (size_t)e * 64 * 64,  W2t, 64 * 64 * 2,  lane, 32);
  for (int i = lane; i < 16 * 128; i += 32) {
    int r = i >> 7, k = i & 127;
    int g = ridx[r];
    Xs[i] = (g >= 0) ? (_Float16)cat[(size_t)g * 128 + k] : (_Float16)0.f;
  }
  wait_async();
  __syncthreads();
  int n = lane & 15;
  int rbase = (lane >> 4) << 3;
  v8f acc[4];
#pragma unroll
  for (int t = 0; t < 4; ++t)
#pragma unroll
    for (int v = 0; v < 8; ++v) acc[t][v] = 0.f;
  for (int kc = 0; kc < 128; kc += 32) {
    v16h a = frag_a(Xs, 128, 0, kc, lane);
#pragma unroll
    for (int t = 0; t < 4; ++t) {
      v16h b = frag_b(W1t, 128, t * 16, kc, lane);
      acc[t] = WMMA16(a, b, acc[t]);
    }
  }
#pragma unroll
  for (int t = 0; t < 4; ++t) {
    int nn = t * 16 + n;
    float bv = B1[e * 64 + nn];
#pragma unroll
    for (int v = 0; v < 8; ++v)
      Hs[(rbase + v) * 64 + nn] = (_Float16)fmaxf(acc[t][v] + bv, 0.f);
  }
  __syncthreads();
#pragma unroll
  for (int t = 0; t < 4; ++t)
#pragma unroll
    for (int v = 0; v < 8; ++v) acc[t][v] = 0.f;
  for (int kc = 0; kc < 64; kc += 32) {
    v16h a = frag_a(Hs, 64, 0, kc, lane);
#pragma unroll
    for (int t = 0; t < 4; ++t) {
      v16h b = frag_b(W2t, 64, t * 16, kc, lane);
      acc[t] = WMMA16(a, b, acc[t]);
    }
  }
#pragma unroll
  for (int t = 0; t < 4; ++t) {
    int nn = t * 16 + n;
    float bv = B2[e * 64 + nn];
#pragma unroll
    for (int v = 0; v < 8; ++v) {
      int g = ridx[rbase + v];
      if (g >= 0) out[(size_t)g * 64 + nn] = acc[t][v] + bv;
    }
  }
}

// ---------------- output head final dot (64 -> 1) ----------------
__global__ void out_head_k(const float* __restrict__ h, const float* __restrict__ W2,
                           float* __restrict__ out, int N) {
  int wave = threadIdx.x >> 5, lane = threadIdx.x & 31;
  int node = blockIdx.x * 8 + wave;
  if (node >= N) return;
  float s = h[(size_t)node * 64 + lane] * W2[lane] +
            h[(size_t)node * 64 + 32 + lane] * W2[32 + lane];
#pragma unroll
  for (int m = 1; m < 32; m <<= 1) s += __shfl_xor(s, m, 32);
  if (lane == 0) out[node] = s;
}

// ---------------- host orchestration ----------------
struct ConvP {
  const float *Wl, *bl, *We, *Wr, *lfg, *lfb, *Wf, *bf, *lpg, *lpb, *Wo1, *bo1, *Wo2, *bo2;
};

extern "C" void kernel_launch(void* const* d_in, const int* in_sizes, int n_in,
                              void* d_out, int out_size, void* d_ws, size_t ws_size,
                              hipStream_t stream) {
  const int NC = in_sizes[0] / 2;
  const int E  = in_sizes[2];
  const int NV = in_sizes[3] / 7;

  const float* consF     = (const float*)d_in[0];
  const int*   eCons     = (const int*)d_in[1];       // edge_indices row 0
  const int*   eVars     = eCons + E;                 // edge_indices row 1
  const float* varF      = (const float*)d_in[3];
  const int*   varColor  = (const int*)d_in[4];
  const int*   consColor = (const int*)d_in[5];
  const float* grpF      = (const float*)d_in[6];
  auto F = [&](int i) { return (const float*)d_in[i]; };

  // params flattened in setup_inputs() insertion order
  const float *grp_lg = F(7), *grp_lb = F(8), *grp_W = F(9), *grp_b = F(10);
  const float *ce_lg = F(11), *ce_lb = F(12), *ce_W1 = F(13), *ce_b1 = F(14),
              *ce_W2 = F(15), *ce_b2 = F(16);
  const float *edge_b = F(18);                        // edge LN over dim-1 => constant b
  const float *ve_lg = F(19), *ve_lb = F(20), *ve_W1 = F(21), *ve_b1 = F(22),
              *ve_W2 = F(23), *ve_b2 = F(24);
  auto mkConv = [&](int b) {
    ConvP c;
    c.Wl = F(b); c.bl = F(b + 1); c.We = F(b + 2); c.Wr = F(b + 3);
    c.lfg = F(b + 4); c.lfb = F(b + 5); c.Wf = F(b + 6); c.bf = F(b + 7);
    c.lpg = F(b + 8); c.lpb = F(b + 9); c.Wo1 = F(b + 10); c.bo1 = F(b + 11);
    c.Wo2 = F(b + 12); c.bo2 = F(b + 13);
    return c;
  };
  ConvP cs[4] = { mkConv(25), mkConv(39), mkConv(53), mkConv(67) };
  const float *col_W1 = F(81), *col_b1 = F(82), *col_W2 = F(83), *col_b2 = F(84);
  const float *o_W1 = F(85), *o_b1 = F(86), *o_W2 = F(87);
  float* out = (float*)d_out;

  // ---- workspace carve (f32 buffers, then ints, then f16 weight pool) ----
  float* wsf = (float*)d_ws;
  size_t o = 0;
  float* cf  = wsf + o; o += (size_t)NC * 64;
  float* vf  = wsf + o; o += (size_t)NV * 64;
  float* ge  = wsf + o; o += (size_t)NV * 64;
  float* Rb  = wsf + o; o += (size_t)NV * 64;
  float* Lb  = wsf + o; o += (size_t)NV * 64;
  float* agg = wsf + o; o += (size_t)NV * 64;
  float* cat = wsf + o; o += (size_t)NV * 128;
  float* hb  = wsf + o; o += (size_t)NV * 64;
  int* idxbuf  = (int*)(wsf + o);
  int* counts  = idxbuf + (NV + 512);
  int* offsets = counts + 32;
  int* padded  = offsets + 33;
  int* cursor  = padded + 33;
  uintptr_t pp = (uintptr_t)(cursor + 32);
  pp = (pp + 15) & ~(uintptr_t)15;
  _Float16* h16 = (_Float16*)pp;
  size_t ho = 0;
  _Float16 *wl16[4], *wr16[4], *wf16[4], *wo1_16[4], *wo2_16[4];
  for (int c = 0; c < 4; ++c) {
    wl16[c]  = h16 + ho; ho += 64 * 64;
    wr16[c]  = h16 + ho; ho += 64 * 64;
    wf16[c]  = h16 + ho; ho += 64 * 64;
    wo1_16[c] = h16 + ho; ho += 64 * 128;
    wo2_16[c] = h16 + ho; ho += 64 * 64;
  }
  _Float16* cew2_16 = h16 + ho; ho += 64 * 64;
  _Float16* vew2_16 = h16 + ho; ho += 64 * 64;
  _Float16* ow1_16  = h16 + ho; ho += 64 * 64;
  _Float16* colW1_16 = h16 + ho; ho += (size_t)32 * 64 * 128;
  _Float16* colW2_16 = h16 + ho; ho += (size_t)32 * 64 * 64;

  // ---- one-time weight transpose/convert to f16 (feeds async DMA staging) ----
  auto prep = [&](const float* W, _Float16* o16, int K, int nE) {
    int tot = 64 * K * nE;
    prep_wt_k<<<(tot + 255) / 256, 256, 0, stream>>>(W, o16, K, nE);
  };
  prep(ce_W2, cew2_16, 64, 1);
  prep(ve_W2, vew2_16, 64, 1);
  prep(o_W1, ow1_16, 64, 1);
  for (int c = 0; c < 4; ++c) {
    prep(cs[c].Wl, wl16[c], 64, 1);
    prep(cs[c].Wr, wr16[c], 64, 1);
    prep(cs[c].Wf, wf16[c], 64, 1);
    prep(cs[c].Wo1, wo1_16[c], 128, 1);
    prep(cs[c].Wo2, wo2_16[c], 64, 1);
  }
  prep(col_W1, colW1_16, 128, 32);
  prep(col_W2, colW2_16, 64, 32);

  auto gemm = [&](const float* X, int K, const _Float16* W16, const float* bias,
                  const float* bias2, const float* b2s, const float* res,
                  float* Y, int N, int relu) {
    gemm_node_k<<<(N + 63) / 64, 128, 0, stream>>>(X, K, W16, bias, bias2, b2s, res, Y, N, relu);
  };

  // ---- node embeddings ----
  grp_emb_k<<<NV, 64, 0, stream>>>(grpF, grp_lg, grp_lb, grp_W, grp_b, ge, NV);
  cons_emb1_k<<<NC, 64, 0, stream>>>(consF, ce_lg, ce_lb, ce_W1, ce_b1, hb, NC);
  gemm(hb, 64, cew2_16, ce_b2, nullptr, nullptr, nullptr, cf, NC, 1);
  var_emb1_k<<<NV, 64, 0, stream>>>(varF, ve_lg, ve_lb, ve_W1, ve_b1, hb, NV);
  gemm(hb, 64, vew2_16, ve_b2, nullptr, nullptr, ge, vf, NV, 1);

  int etiles = (E + 15) / 16;
  auto bip = [&](const float* left, const float* right, const int* src, const int* dst,
                 int n_left, int n_right, int c) {
    const ConvP& p = cs[c];
    gemm(right, 64, wl16[c], p.bl, p.We, edge_b, nullptr, Rb, n_right, 0);
    gemm(left, 64, wr16[c], nullptr, nullptr, nullptr, nullptr, Lb, n_left, 0);
    int na = n_right * 64;
    zero_f_k<<<(na + 255) / 256, 256, 0, stream>>>(agg, na);
    edge_msg_k<<<(etiles + 3) / 4, 128, 0, stream>>>(Rb, Lb, src, dst, p.lfg, p.lfb,
                                                     wf16[c], p.bf, agg, E);
    postcat_k<<<(n_right + 7) / 8, 256, 0, stream>>>(agg, right, p.lpg, p.lpb, cat, n_right);
  };
  auto colorMlp = [&](const int* color, int n, float* dstF) {
    fill_i32_k<<<1, 32, 0, stream>>>(counts, 0, 32);
    int np = n + 512;
    fill_i32_k<<<(np + 255) / 256, 256, 0, stream>>>(idxbuf, -1, np);
    moe_count_k<<<(n + 255) / 256, 256, 0, stream>>>(color, counts, n);
    moe_scan_k<<<1, 32, 0, stream>>>(counts, offsets, padded, cursor);
    moe_scatter_k<<<(n + 255) / 256, 256, 0, stream>>>(color, cursor, idxbuf, n);
    int maxTiles = (n + 480 + 15) / 16;                 // worst-case padded tile count
    color_gemm_k<<<maxTiles, 32, 0, stream>>>(cat, idxbuf, padded,
                                              colW1_16, col_b1, colW2_16, col_b2, dstF);
  };

  // ---- conv 1: v -> c, color MLP ----
  bip(vf, cf, eVars, eCons, NV, NC, 0);
  colorMlp(consColor, NC, cf);
  // ---- conv 2: c -> v, color MLP ----
  bip(cf, vf, eCons, eVars, NC, NV, 1);
  colorMlp(varColor, NV, vf);
  // ---- conv 3: v -> c, plain MLP ----
  bip(vf, cf, eVars, eCons, NV, NC, 2);
  gemm(cat, 128, wo1_16[2], cs[2].bo1, nullptr, nullptr, nullptr, hb, NC, 1);
  gemm(hb, 64, wo2_16[2], cs[2].bo2, nullptr, nullptr, nullptr, cf, NC, 0);
  // ---- conv 4: c -> v, plain MLP ----
  bip(cf, vf, eCons, eVars, NC, NV, 3);
  gemm(cat, 128, wo1_16[3], cs[3].bo1, nullptr, nullptr, nullptr, hb, NV, 1);
  gemm(hb, 64, wo2_16[3], cs[3].bo2, nullptr, nullptr, nullptr, vf, NV, 0);
  // ---- output head ----
  gemm(vf, 64, ow1_16, o_b1, nullptr, nullptr, nullptr, hb, NV, 1);
  out_head_k<<<(NV + 7) / 8, 256, 0, stream>>>(hb, o_W2, out, NV);
}
